// GatedCrossAttentionBlock_8847632630098
// MI455X (gfx1250) — matrix operand order repaired and checked by reading, not compiled, not run on hardware
//
#include <hip/hip_runtime.h>
#include <hip/hip_bf16.h>
#include <math.h>

typedef __bf16 bf16;
typedef __attribute__((ext_vector_type(16))) __bf16 bf16x16;
typedef __attribute__((ext_vector_type(8)))  __bf16 bf16x8;
typedef __attribute__((ext_vector_type(8)))  float  f32x8;

union Frag { bf16x16 v; bf16x8 h[2]; };

__device__ __forceinline__ f32x8 wmma_bf16(const Frag& a, const Frag& b, f32x8 c) {
  // (neg_a, A, neg_b, B, c_mod, C, reuse_a, reuse_b)
  return __builtin_amdgcn_wmma_f32_16x16x32_bf16(false, a.v, false, b.v, (short)0, c,
                                                 false, false);
}

// ---------------------------------------------------------------------------
// Weight transpose + fp32->bf16 cast:  Wt[n*K + k] = W[k*N + n]
// ---------------------------------------------------------------------------
__global__ __launch_bounds__(256) void transpose_cast_kernel(
    const float* __restrict__ W, bf16* __restrict__ Wt, int K, int N) {
  size_t total = (size_t)K * N;
  for (size_t i = (size_t)blockIdx.x * blockDim.x + threadIdx.x; i < total;
       i += (size_t)gridDim.x * blockDim.x) {
    size_t n = i / (size_t)K, k = i % (size_t)K;
    Wt[i] = (bf16)W[k * (size_t)N + n];
  }
}

__global__ __launch_bounds__(256) void cast_kernel(
    const float* __restrict__ in, bf16* __restrict__ out, int n) {
  int i = blockIdx.x * blockDim.x + threadIdx.x;
  if (i < n) out[i] = (bf16)in[i];
}

// vT[((b*16+h)*64 + d)*64 + j] = v[b, j, h, d]  (v = cols [1024,2048) of kv)
__global__ __launch_bounds__(256) void build_vT_kernel(
    const bf16* __restrict__ kv, bf16* __restrict__ vT) {
  int i = blockIdx.x * blockDim.x + threadIdx.x;   // 262144 total
  int j = i & 63, d = (i >> 6) & 63, h = (i >> 12) & 15, b = i >> 16;
  vT[i] = kv[(size_t)(b * 64 + j) * 2048 + 1024 + h * 64 + d];
}

// ---------------------------------------------------------------------------
// Row LayerNorm (D = 2048), fp32 in -> bf16 out. One block per row.
// ---------------------------------------------------------------------------
__global__ __launch_bounds__(256) void layernorm_kernel(
    const float* __restrict__ x, const float* __restrict__ g,
    const float* __restrict__ bta, bf16* __restrict__ out, int D) {
  __shared__ float r1[256];
  __shared__ float r2[256];
  const float* row = x + (size_t)blockIdx.x * D;
  float s = 0.f, s2 = 0.f;
  for (int i = threadIdx.x; i < D; i += 256) { float v = row[i]; s += v; s2 += v * v; }
  r1[threadIdx.x] = s; r2[threadIdx.x] = s2;
  __syncthreads();
  for (int o = 128; o > 0; o >>= 1) {
    if ((int)threadIdx.x < o) {
      r1[threadIdx.x] += r1[threadIdx.x + o];
      r2[threadIdx.x] += r2[threadIdx.x + o];
    }
    __syncthreads();
  }
  float mu  = r1[0] / (float)D;
  float var = r2[0] / (float)D - mu * mu;
  float rstd = rsqrtf(var + 1e-5f);
  bf16* orow = out + (size_t)blockIdx.x * D;
  for (int i = threadIdx.x; i < D; i += 256)
    orow[i] = (bf16)((row[i] - mu) * rstd * g[i] + bta[i]);
}

// ---------------------------------------------------------------------------
// Tiled bf16 GEMM: C[M,N] = epilogue(alpha * A[M,K] @ Bt[N,K]^T)
//   A row-major (lda), Bt is B pre-transposed: row n = column n of B (ldb=K)
//   wave tile = 32 x 64 (2 M-tiles x 4 N-tiles = 8 accumulators); each B
//   fragment is amortized over 2 WMMAs -> 12 b128 loads per 8 WMMAs.
//   block = 8 waves -> 256 x 64 C tile.
// MODE 0: bf16 store            MODE 1: bf16 store of exact GELU
// MODE 2: f32 store  resid[idx] + tanh(gate)*v   (resid may alias Cout)
// ---------------------------------------------------------------------------
template <int MODE>
__global__ __launch_bounds__(256) void gemm_kernel(
    const bf16* __restrict__ A, const bf16* __restrict__ Bt,
    int K, int lda, int ldb, int ldc, float alpha,
    void* Cout, const float* resid, const float* gate) {
  int wave = threadIdx.x >> 5, lane = threadIdx.x & 31;
  int l = lane & 15, half = lane >> 4;
  int row0 = blockIdx.y * 256 + wave * 32;   // M-tiles at row0 and row0+16
  int n0 = blockIdx.x * 64;

  f32x8 z = {0.f, 0.f, 0.f, 0.f, 0.f, 0.f, 0.f, 0.f};
  f32x8 acc[2][4] = {{z, z, z, z}, {z, z, z, z}};

  const bf16* arow0 = A + (size_t)(row0 + l) * lda + half * 8;
  const bf16* arow1 = arow0 + (size_t)16 * lda;
  for (int k0 = 0; k0 < K; k0 += 32) {
    Frag a0, a1;
    a0.h[0] = *(const bf16x8*)(arow0 + k0);
    a0.h[1] = *(const bf16x8*)(arow0 + k0 + 16);
    a1.h[0] = *(const bf16x8*)(arow1 + k0);
    a1.h[1] = *(const bf16x8*)(arow1 + k0 + 16);
#pragma unroll
    for (int t = 0; t < 4; ++t) {
      const bf16* bp = Bt + (size_t)(n0 + t * 16 + l) * ldb + k0 + half * 8;
      Frag b;
      b.h[0] = *(const bf16x8*)(bp);
      b.h[1] = *(const bf16x8*)(bp + 16);
      acc[0][t] = wmma_bf16(a0, b, acc[0][t]);
      acc[1][t] = wmma_bf16(a1, b, acc[1][t]);
    }
  }

  float tg = (MODE == 2) ? tanhf(gate[0]) : 0.f;
#pragma unroll
  for (int m = 0; m < 2; ++m) {
#pragma unroll
    for (int t = 0; t < 4; ++t) {
#pragma unroll
      for (int r = 0; r < 8; ++r) {
        int row = row0 + m * 16 + half * 8 + r;
        int col = n0 + t * 16 + l;
        size_t idx = (size_t)row * ldc + col;
        float v = acc[m][t][r] * alpha;
        if (MODE == 0) {
          ((bf16*)Cout)[idx] = (bf16)v;
        } else if (MODE == 1) {
          float ge = 0.5f * v * (1.0f + erff(v * 0.70710678118654752f));
          ((bf16*)Cout)[idx] = (bf16)ge;
        } else {
          ((float*)Cout)[idx] = resid[idx] + tg * v;
        }
      }
    }
  }
}

// ---------------------------------------------------------------------------
// Fused masked cross-attention: one wave per (b, h, 16-row tile of queries).
//   S = Q Kt (WMMA) -> mask -> softmax (shfl within 16-lane half) ->
//   P staged via LDS (C-layout -> A-layout) -> O = P V (WMMA)
//   q: (B*N, 1024) bf16 (scale pre-applied)   kv: (B*64, 2048) bf16
//   vT: (B,H,64d,64j) bf16                    O: (B*N, 1024) bf16
// ---------------------------------------------------------------------------
__global__ __launch_bounds__(256) void attn_kernel(
    const bf16* __restrict__ q, const bf16* __restrict__ kv,
    const bf16* __restrict__ vT, const unsigned char* __restrict__ mask,
    bf16* __restrict__ O) {
  __shared__ bf16 lds[8][16 * 64];
  int wave = threadIdx.x >> 5, lane = threadIdx.x & 31;
  int l = lane & 15, half = lane >> 4;
  int tile = blockIdx.x * 8 + wave;       // 0 .. 8191
  int i0 = (tile & 127) << 4;
  int h = (tile >> 7) & 15;
  int b = tile >> 11;

  f32x8 z = {0.f, 0.f, 0.f, 0.f, 0.f, 0.f, 0.f, 0.f};
  f32x8 s[4] = {z, z, z, z};

  // ---- S = Q @ K^T   (K dim = head dim 64) ----
  const bf16* qrow = q + (size_t)(b * 2048 + i0 + l) * 1024 + h * 64 + half * 8;
#pragma unroll
  for (int k0 = 0; k0 < 64; k0 += 32) {
    Frag a;
    a.h[0] = *(const bf16x8*)(qrow + k0);
    a.h[1] = *(const bf16x8*)(qrow + k0 + 16);
#pragma unroll
    for (int t = 0; t < 4; ++t) {
      const bf16* bp = kv + (size_t)(b * 64 + t * 16 + l) * 2048 + h * 64 + k0 + half * 8;
      Frag bb;
      bb.h[0] = *(const bf16x8*)(bp);
      bb.h[1] = *(const bf16x8*)(bp + 16);
      s[t] = wmma_bf16(a, bb, s[t]);
    }
  }

  // ---- mask (branchless; column j = t*16 + l) ----
#pragma unroll
  for (int t = 0; t < 4; ++t) {
    float pen = (mask[b * 64 + t * 16 + l] != 0) ? 0.0f : -3.0e38f;
#pragma unroll
    for (int r = 0; r < 8; ++r) s[t][r] = s[t][r] + pen;
  }

  // ---- softmax over 64 columns per row (row = half*8 + r) ----
  float mx[8], sm[8];
#pragma unroll
  for (int r = 0; r < 8; ++r) {
    float m = fmaxf(fmaxf(s[0][r], s[1][r]), fmaxf(s[2][r], s[3][r]));
    m = fmaxf(m, __shfl_xor(m, 1, 32));
    m = fmaxf(m, __shfl_xor(m, 2, 32));
    m = fmaxf(m, __shfl_xor(m, 4, 32));
    m = fmaxf(m, __shfl_xor(m, 8, 32));
    mx[r] = m;
    sm[r] = 0.f;
  }
#pragma unroll
  for (int t = 0; t < 4; ++t)
#pragma unroll
    for (int r = 0; r < 8; ++r) {
      float p = __expf(s[t][r] - mx[r]);
      s[t][r] = p;
      sm[r] += p;
    }
#pragma unroll
  for (int r = 0; r < 8; ++r) {
    float v = sm[r];
    v += __shfl_xor(v, 1, 32);
    v += __shfl_xor(v, 2, 32);
    v += __shfl_xor(v, 4, 32);
    v += __shfl_xor(v, 8, 32);
    sm[r] = 1.0f / v;
  }

  // ---- P: C-layout -> LDS (row-major 16x64) -> A-layout ----
#pragma unroll
  for (int t = 0; t < 4; ++t)
#pragma unroll
    for (int r = 0; r < 8; ++r)
      lds[wave][(half * 8 + r) * 64 + t * 16 + l] = (bf16)(s[t][r] * sm[r]);
  // LDS ops are in-order within a wave; compiler inserts s_wait_dscnt for reads.

  // ---- O = P @ V   (K dim = 64 latents) ----
  f32x8 o[4] = {z, z, z, z};
  const bf16* prow = &lds[wave][l * 64 + half * 8];
#pragma unroll
  for (int k0 = 0; k0 < 64; k0 += 32) {
    Frag a;
    a.h[0] = *(const bf16x8*)(prow + k0);
    a.h[1] = *(const bf16x8*)(prow + k0 + 16);
#pragma unroll
    for (int t = 0; t < 4; ++t) {
      const bf16* bp = vT + (size_t)((b * 16 + h) * 64 + t * 16 + l) * 64 + k0 + half * 8;
      Frag bb;
      bb.h[0] = *(const bf16x8*)(bp);
      bb.h[1] = *(const bf16x8*)(bp + 16);
      o[t] = wmma_bf16(a, bb, o[t]);
    }
  }
#pragma unroll
  for (int t = 0; t < 4; ++t)
#pragma unroll
    for (int r = 0; r < 8; ++r)
      O[(size_t)(b * 2048 + i0 + half * 8 + r) * 1024 + h * 64 + t * 16 + l] = (bf16)o[t][r];
}

// ---------------------------------------------------------------------------
extern "C" void kernel_launch(void* const* d_in, const int* in_sizes, int n_in,
                              void* d_out, int out_size, void* d_ws, size_t ws_size,
                              hipStream_t stream) {
  (void)in_sizes; (void)n_in; (void)out_size; (void)ws_size;
  const float* x         = (const float*)d_in[0];   // (4,2048,2048)
  const float* media     = (const float*)d_in[1];   // (4,64,1024)
  const unsigned char* mask = (const unsigned char*)d_in[2]; // (4,64) bool
  const float* ln_q_g    = (const float*)d_in[3];
  const float* ln_q_b    = (const float*)d_in[4];
  const float* Wq        = (const float*)d_in[5];   // (2048,1024)
  const float* Wkv       = (const float*)d_in[6];   // (1024,2048)
  const float* Wo        = (const float*)d_in[7];   // (1024,2048)
  const float* attn_gate = (const float*)d_in[8];
  const float* ln_ff_g   = (const float*)d_in[9];
  const float* ln_ff_b   = (const float*)d_in[10];
  const float* W1        = (const float*)d_in[11];  // (2048,8192)
  const float* W2        = (const float*)d_in[12];  // (8192,2048)
  const float* ff_gate   = (const float*)d_in[13];
  float* out = (float*)d_out;                       // (4,2048,2048) fp32

  char* base = (char*)d_ws;
  size_t off = 0;
  auto take = [&](size_t bytes) -> char* {
    char* r = base + off;
    off += (bytes + 255) & ~(size_t)255;
    return r;
  };
  bf16* Wq_t   = (bf16*)take((size_t)1024 * 2048 * 2);  // (N=1024, K=2048)
  bf16* Wkv_t  = (bf16*)take((size_t)2048 * 1024 * 2);  // (N=2048, K=1024)
  bf16* Wo_t   = (bf16*)take((size_t)2048 * 1024 * 2);  // (N=2048, K=1024)
  bf16* W1_t   = (bf16*)take((size_t)8192 * 2048 * 2);  // (N=8192, K=2048)
  bf16* W2_t   = (bf16*)take((size_t)2048 * 8192 * 2);  // (N=2048, K=8192)
  bf16* qn     = (bf16*)take((size_t)8192 * 2048 * 2);
  bf16* qb     = (bf16*)take((size_t)8192 * 1024 * 2);
  bf16* mediab = (bf16*)take((size_t)256 * 1024 * 2);
  bf16* kvb    = (bf16*)take((size_t)256 * 2048 * 2);
  bf16* vTb    = (bf16*)take((size_t)262144 * 2);
  bf16* Ob     = (bf16*)take((size_t)8192 * 1024 * 2);
  bf16* hb     = (bf16*)take((size_t)8192 * 2048 * 2);
  bf16* gb     = (bf16*)take((size_t)8192 * 8192 * 2);

  // Weight prep (cheap vs GEMMs; bf16 weight set lives in 192MB L2)
  transpose_cast_kernel<<<2048, 256, 0, stream>>>(Wq,  Wq_t,  2048, 1024);
  transpose_cast_kernel<<<2048, 256, 0, stream>>>(Wkv, Wkv_t, 1024, 2048);
  transpose_cast_kernel<<<2048, 256, 0, stream>>>(Wo,  Wo_t,  1024, 2048);
  transpose_cast_kernel<<<4096, 256, 0, stream>>>(W1,  W1_t,  2048, 8192);
  transpose_cast_kernel<<<4096, 256, 0, stream>>>(W2,  W2_t,  8192, 2048);

  // Attention branch
  layernorm_kernel<<<8192, 256, 0, stream>>>(x, ln_q_g, ln_q_b, qn, 2048);
  cast_kernel<<<1024, 256, 0, stream>>>(media, mediab, 262144);
  // q = LN(x) @ Wq * 1/sqrt(64)   (M=8192 -> 32 row-blocks of 256)
  gemm_kernel<0><<<dim3(16, 32), 256, 0, stream>>>(qn, Wq_t, 2048, 2048, 2048, 1024,
                                                   0.125f, qb, nullptr, nullptr);
  // kv = media @ Wkv              (M=256 -> 1 row-block)
  gemm_kernel<0><<<dim3(32, 1), 256, 0, stream>>>(mediab, Wkv_t, 1024, 1024, 1024, 2048,
                                                  1.0f, kvb, nullptr, nullptr);
  build_vT_kernel<<<1024, 256, 0, stream>>>(kvb, vTb);
  attn_kernel<<<1024, 256, 0, stream>>>(qb, kvb, vTb, mask, Ob);
  // x2 = x + tanh(attn_gate) * (O @ Wo)   -> d_out (fp32)
  gemm_kernel<2><<<dim3(32, 32), 256, 0, stream>>>(Ob, Wo_t, 1024, 1024, 1024, 2048,
                                                   1.0f, out, x, attn_gate);

  // FFN branch
  layernorm_kernel<<<8192, 256, 0, stream>>>(out, ln_ff_g, ln_ff_b, hb, 2048);
  // g = gelu(h @ W1)
  gemm_kernel<1><<<dim3(128, 32), 256, 0, stream>>>(hb, W1_t, 2048, 2048, 2048, 8192,
                                                    1.0f, gb, nullptr, nullptr);
  // out = x2 + tanh(ff_gate) * (g @ W2)   (in-place on d_out)
  gemm_kernel<2><<<dim3(32, 32), 256, 0, stream>>>(gb, W2_t, 8192, 8192, 8192, 2048,
                                                   1.0f, out, out, ff_gate);
}
